// MultiHeadAttentionBlock_1382979469695
// MI455X (gfx1250) — matrix operand order repaired and checked by reading, not compiled, not run on hardware
//
#include <hip/hip_runtime.h>
#include <hip/hip_bf16.h>

// ---------------------------------------------------------------------------
// Problem constants (match reference)
// ---------------------------------------------------------------------------
#define B_   4
#define S_   2048
#define D_   1024
#define H_   16
#define DH_  64
#define HD_  (H_ * DH_)   // 1024

typedef __attribute__((ext_vector_type(16))) __bf16 v16bf;
typedef __attribute__((ext_vector_type(8)))  float  v8f;
typedef __attribute__((ext_vector_type(4)))  unsigned int v4u_;
typedef __attribute__((ext_vector_type(8)))  int v8i_;
typedef __attribute__((ext_vector_type(4)))  int v4i_;

#if __has_builtin(__builtin_amdgcn_tensor_load_to_lds)
#define ATTN_TDM 1
#else
#define ATTN_TDM 0
#endif

union AFrag {              // 16x32 bf16 A-matrix fragment (8 VGPRs/lane)
    v16bf v;
    unsigned int u[8];
};

// A-fragment K-base for packed pair i (0..7), contiguous order matching B:
// K = 2i + 16*half
__device__ __forceinline__ int kbase(int i, int half) {
    return i * 2 + half * 16;
}

__device__ __forceinline__ v8f wmma_bf16(v16bf a, v16bf b, v8f c) {
    return __builtin_amdgcn_wmma_f32_16x16x32_bf16(
        /*neg_a=*/false, a, /*neg_b=*/false, b,
        /*c_mod=*/(short)0, c, /*reuse_a=*/false, /*reuse_b=*/false);
}

__device__ __forceinline__ float rmax16(float v) {
    v = fmaxf(v, __shfl_xor(v, 1, 32));
    v = fmaxf(v, __shfl_xor(v, 2, 32));
    v = fmaxf(v, __shfl_xor(v, 4, 32));
    v = fmaxf(v, __shfl_xor(v, 8, 32));
    return v;
}
__device__ __forceinline__ float rsum16(float v) {
    v += __shfl_xor(v, 1, 32);
    v += __shfl_xor(v, 2, 32);
    v += __shfl_xor(v, 4, 32);
    v += __shfl_xor(v, 8, 32);
    return v;
}

// Async global -> LDS copy of 16 bytes (CDNA5, tracked by ASYNCcnt).
__device__ __forceinline__ void async_copy_b128(void* lds_ptr, const void* gptr) {
    unsigned lds            = (unsigned)(unsigned long long)lds_ptr;
    unsigned long long addr = (unsigned long long)gptr;
    asm volatile("global_load_async_to_lds_b128 %0, %1, off"
                 :: "v"(lds), "v"(addr) : "memory");
}
__device__ __forceinline__ void wait_asynccnt_0() {
    asm volatile("s_wait_asynccnt 0x0" ::: "memory");
}
__device__ __forceinline__ void wait_asynccnt_3() {
    asm volatile("s_wait_asynccnt 0x3" ::: "memory");
}

__device__ __forceinline__ void wait_tensorcnt_0() {
#if __has_builtin(__builtin_amdgcn_s_wait_tensorcnt)
    __builtin_amdgcn_s_wait_tensorcnt((short)0);
#else
    asm volatile("s_wait_tensorcnt 0x0" ::: "memory");
#endif
}
__device__ __forceinline__ void wait_tensorcnt_2() {
#if __has_builtin(__builtin_amdgcn_s_wait_tensorcnt)
    __builtin_amdgcn_s_wait_tensorcnt((short)2);
#else
    asm volatile("s_wait_tensorcnt 0x2" ::: "memory");
#endif
}

#if ATTN_TDM
// TDM 2D tile load: global (bf16, data_size=2B) -> LDS.
// D# per ISA 8.3/8.4: group0 = {count|flags, lds_addr, gaddr_lo, gaddr_hi|type},
// group1 = {mask/data_size, dims/tiles/strides}. 2D only: groups 2/3 zeroed.
__device__ __forceinline__ void tdm_load_2d(unsigned lds_addr,
                                            unsigned long long gaddr,
                                            unsigned tile_d0, unsigned tile_d1,
                                            unsigned tensor_d0, unsigned tensor_d1,
                                            unsigned stride0) {
    v4u_ g0;
    g0.x = 1u;                                     // count=1, user mode
    g0.y = lds_addr;                               // LDS byte address
    g0.z = (unsigned)gaddr;                        // global addr [31:0]
    g0.w = (unsigned)(gaddr >> 32) | (2u << 30);   // addr [56:32] | type=2
    v8i_ g1;
    g1[0] = (int)(1u << 16);                       // data_size=1 (2 bytes)
    g1[1] = (int)(tensor_d0 << 16);                // tensor_dim0 [63:48]
    g1[2] = (int)((tensor_d0 >> 16) | (tensor_d1 << 16));   // [79:64],[95:80]
    g1[3] = (int)(((tensor_d1 >> 16) & 0xFFFFu) | (tile_d0 << 16)); // tile_dim0
    g1[4] = (int)(tile_d1 & 0xFFFFu);              // tile_dim1 ; tile_dim2=0
    g1[5] = (int)stride0;                          // tensor_dim0_stride lo32
    g1[6] = 0;
    g1[7] = 0;
    v4i_ gz = {0, 0, 0, 0};
#if defined(__clang_major__) && (__clang_major__ >= 23)
    v8i_ gz8 = {0, 0, 0, 0, 0, 0, 0, 0};
    __builtin_amdgcn_tensor_load_to_lds(g0, g1, gz, gz, gz8, 0);
#else
    __builtin_amdgcn_tensor_load_to_lds(g0, g1, gz, gz, 0);
#endif
}
#endif

// ---------------------------------------------------------------------------
// Kernel 0: transpose + convert f32 [batch][R][C] -> bf16 [batch][C][R]
// ---------------------------------------------------------------------------
__global__ __launch_bounds__(256)
void k_prep_transpose(const float* __restrict__ src, __bf16* __restrict__ dst,
                      int R, int C) {
    int batch = blockIdx.y;
    src += (size_t)batch * R * C;
    dst += (size_t)batch * R * C;
    int idx = blockIdx.x * 256 + threadIdx.x;
    if (idx < R * C) {
        int c = idx / R;
        int r = idx - c * R;
        dst[idx] = (__bf16)src[(size_t)r * C + c];
    }
}

// ---------------------------------------------------------------------------
// Kernel 1: fused QKV projection.
//   x[B,S,D] f32  x  Wt[H,DH,D] bf16 (pre-transposed)  -> per head 16x64 tiles
//   Q,K stored [B,H,S,DH] bf16 ; V stored transposed [B,H,DH,S] bf16
// ---------------------------------------------------------------------------
__global__ __launch_bounds__(256)
void k_qkv(const float* __restrict__ x,
           const __bf16* __restrict__ Wqt, const __bf16* __restrict__ Wkt,
           const __bf16* __restrict__ Wvt,
           const float* __restrict__ bq, const float* __restrict__ bk,
           const float* __restrict__ bv,
           __bf16* __restrict__ Q, __bf16* __restrict__ K,
           __bf16* __restrict__ Vt) {
    __shared__ __bf16 xs[16 * D_];     // 32 KB, row-major [m][d]

    const int tid  = threadIdx.x;
    const int b    = blockIdx.x / (S_ / 16);
    const int s0   = (blockIdx.x % (S_ / 16)) * 16;

    // Stage x tile (16 x 1024 f32 -> bf16). 16384 elems, 4 per thread/iter.
    #pragma unroll
    for (int it = 0; it < 16; ++it) {
        int idx4 = (tid + it * 256) * 4;           // multiple of 4
        int m = idx4 >> 10, d = idx4 & (D_ - 1);
        float4 f = *(const float4*)(x + ((size_t)(b * S_ + s0 + m) * D_ + d));
        union { __bf16 h[4]; uint2 u; } pk;
        pk.h[0] = (__bf16)f.x; pk.h[1] = (__bf16)f.y;
        pk.h[2] = (__bf16)f.z; pk.h[3] = (__bf16)f.w;
        *(uint2*)(xs + idx4) = pk.u;
    }
    __syncthreads();

    const int wave = tid >> 5;
    const int lane = tid & 31;
    const int n    = lane & 15;
    const int half = lane >> 4;

    // 48 tasks = {q,k,v} x 16 heads ; 6 per wave
    for (int task = wave; task < 48; task += 8) {
        const int mat = task / H_;     // 0=q 1=k 2=v
        const int h   = task % H_;
        const __bf16* Wt  = (mat == 0 ? Wqt : (mat == 1 ? Wkt : Wvt))
                            + (size_t)h * DH_ * D_;
        const float*  bia = (mat == 0 ? bq : (mat == 1 ? bk : bv)) + h * DH_;

        v8f acc[4];
        #pragma unroll
        for (int nt = 0; nt < 4; ++nt) {
            v8f z = {0.f,0.f,0.f,0.f,0.f,0.f,0.f,0.f};
            acc[nt] = z;
        }

        for (int kc = 0; kc < D_ / 32; ++kc) {
            AFrag a;
            #pragma unroll
            for (int i = 0; i < 8; ++i)
                a.u[i] = *(const unsigned int*)(xs + n * D_ + kc * 32 + kbase(i, half));
            #pragma unroll
            for (int nt = 0; nt < 4; ++nt) {
                int dh = nt * 16 + n;
                v16bf bf = *(const v16bf*)(Wt + (size_t)dh * D_ + kc * 32 + half * 16);
                acc[nt] = wmma_bf16(a.v, bf, acc[nt]);
            }
        }

        #pragma unroll
        for (int nt = 0; nt < 4; ++nt) {
            int dh = nt * 16 + n;
            float bb = bia[dh];
            #pragma unroll
            for (int r = 0; r < 8; ++r) {
                float val  = acc[nt][r] + bb;
                int   srow = s0 + r + 8 * half;
                if (mat == 0)
                    Q[((size_t)(b * H_ + h) * S_ + srow) * DH_ + dh] = (__bf16)val;
                else if (mat == 1)
                    K[((size_t)(b * H_ + h) * S_ + srow) * DH_ + dh] = (__bf16)val;
                else
                    Vt[((size_t)(b * H_ + h) * DH_ + dh) * S_ + srow] = (__bf16)val;
            }
        }
    }
}

// ---------------------------------------------------------------------------
// Kernel 2: flash attention. Block = 8 waves sharing (b,h); wave w owns the
// 16-row s-tile s0 = sg*128 + w*16. K/V chunks (32 columns of t) are staged
// block-cooperatively into LDS, double-buffered:
//   - TDM path: wave 0 issues one tensor_load_to_lds per K / V chunk
//     (TENSORcnt), DMA overlaps with WMMA + softmax of the current chunk.
//   - Fallback: per-thread async global->LDS b128 copies (ASYNCcnt).
// ---------------------------------------------------------------------------
__global__ __launch_bounds__(256)
void k_attn(const __bf16* __restrict__ Q, const __bf16* __restrict__ K,
            const __bf16* __restrict__ Vt, __bf16* __restrict__ cat) {
    __shared__ __bf16 Ks[2][32 * DH_];      // 2 x 4 KB : K rows t0..t0+31
    __shared__ __bf16 Vs[2][DH_ * 32];      // 2 x 4 KB : V^T [dh][t-rel]
    __shared__ __bf16 plds[8 * 16 * 32];    // per-wave P transpose buffer

    const int tid  = threadIdx.x;
    const int wave = tid >> 5;
    const int lane = tid & 31;
    const int n    = lane & 15;
    const int half = lane >> 4;

    const int bh = blockIdx.x / 16;        // b*H + h
    const int sg = blockIdx.x % 16;
    const int b  = bh / H_;
    const int h  = bh % H_;
    const int s0 = sg * 128 + wave * 16;

    const __bf16* Qp  = Q  + (size_t)bh * S_ * DH_;
    const __bf16* Kp  = K  + (size_t)bh * S_ * DH_;
    const __bf16* Vtp = Vt + (size_t)bh * DH_ * S_;
    __bf16* pb = plds + wave * (16 * 32);

    // ---- Q A-fragments (K-dim = e: 0..31 and 32..63), loaded once
    AFrag qa0, qa1;
    #pragma unroll
    for (int i = 0; i < 8; ++i) {
        const __bf16* qrow = Qp + (size_t)(s0 + n) * DH_;
        qa0.u[i] = *(const unsigned int*)(qrow +      kbase(i, half));
        qa1.u[i] = *(const unsigned int*)(qrow + 32 + kbase(i, half));
    }

    // ---- stage one 32-column chunk into LDS buffer `buf`
    auto issue_chunk = [&](int tc, int buf) {
        const int t0 = tc * 32;
#if ATTN_TDM
        if (wave == 0) {
            // K rows t0..t0+31: 4 KB contiguous ([S][64] row-major) -> 1D tile
            tdm_load_2d((unsigned)(unsigned long long)(void*)&Ks[buf][0],
                        (unsigned long long)(const void*)(Kp + (size_t)t0 * DH_),
                        /*tile_d0=*/32 * DH_, /*tile_d1=*/1,
                        /*tensor_d0=*/32 * DH_, /*tensor_d1=*/1,
                        /*stride0=*/32 * DH_);
            // V^T: 64 rows (dh) x 32 cols (t-rel), row stride S_ elements
            tdm_load_2d((unsigned)(unsigned long long)(void*)&Vs[buf][0],
                        (unsigned long long)(const void*)(Vtp + t0),
                        /*tile_d0=*/32, /*tile_d1=*/DH_,
                        /*tensor_d0=*/S_, /*tensor_d1=*/DH_,
                        /*stride0=*/S_);
        }
#else
        #pragma unroll
        for (int it = 0; it < 2; ++it) {
            int idx8 = (tid + it * 256) * 8;
            async_copy_b128(&Ks[buf][idx8], Kp + (size_t)t0 * DH_ + idx8);
        }
        {
            int dh = tid >> 2, part = (tid & 3) * 8;
            async_copy_b128(&Vs[buf][dh * 32 + part],
                            Vtp + (size_t)dh * S_ + t0 + part);
        }
#endif
    };

    float rowM[8], rowL[8];
    v8f o[4];
    #pragma unroll
    for (int r = 0; r < 8; ++r) { rowM[r] = -3.0e38f; rowL[r] = 0.f; }
    #pragma unroll
    for (int nt = 0; nt < 4; ++nt) {
        v8f z = {0.f,0.f,0.f,0.f,0.f,0.f,0.f,0.f};
        o[nt] = z;
    }
    const float scale = 0.125f;   // 1/sqrt(64)
    const int   NC    = S_ / 32;  // 64 chunks

    issue_chunk(0, 0);

    for (int tc = 0; tc < NC; ++tc) {
        const int buf = tc & 1;
        if (tc + 1 < NC) {
            issue_chunk(tc + 1, buf ^ 1);
#if ATTN_TDM
            wait_tensorcnt_2();            // chunk tc landed; tc+1 in flight
#else
            wait_asynccnt_3();
#endif
        } else {
#if ATTN_TDM
            wait_tensorcnt_0();
#else
            wait_asynccnt_0();
#endif
        }
        __syncthreads();                   // all waves see staged data

        const __bf16* ks = Ks[buf];
        const __bf16* vs = Vs[buf];

        // ---- scores: two 16x16 tiles (cols t-rel 0..15 and 16..31)
        v8f s1 = {0.f,0.f,0.f,0.f,0.f,0.f,0.f,0.f};
        v8f s2 = s1;
        {
            v16bf b00 = *(const v16bf*)(ks + n * DH_ + half * 16);
            v16bf b01 = *(const v16bf*)(ks + n * DH_ + 32 + half * 16);
            v16bf b10 = *(const v16bf*)(ks + (16 + n) * DH_ + half * 16);
            v16bf b11 = *(const v16bf*)(ks + (16 + n) * DH_ + 32 + half * 16);
            s1 = wmma_bf16(qa0.v, b00, s1);
            s1 = wmma_bf16(qa1.v, b01, s1);
            s2 = wmma_bf16(qa0.v, b10, s2);
            s2 = wmma_bf16(qa1.v, b11, s2);
        }

        // ---- online softmax update (row-wise over 32 cols)
        float fac[8];
        #pragma unroll
        for (int r = 0; r < 8; ++r) {
            float a = s1[r] * scale;
            float c = s2[r] * scale;
            float mx = rmax16(fmaxf(a, c));
            float nm = fmaxf(rowM[r], mx);
            float p1 = __expf(a - nm);
            float p2 = __expf(c - nm);
            s1[r] = p1; s2[r] = p2;
            float f = __expf(rowM[r] - nm);
            rowM[r] = nm;
            float sm = rsum16(p1 + p2);
            rowL[r] = rowL[r] * f + sm;
            fac[r] = f;
        }
        #pragma unroll
        for (int nt = 0; nt < 4; ++nt)
            #pragma unroll
            for (int r = 0; r < 8; ++r)
                o[nt][r] *= fac[r];

        // ---- transpose P through LDS: C/D layout -> A layout (as bf16)
        #pragma unroll
        for (int r = 0; r < 8; ++r) {
            int mrow = r + 8 * half;
            pb[mrow * 32 + n]      = (__bf16)s1[r];
            pb[mrow * 32 + 16 + n] = (__bf16)s2[r];
        }
        AFrag pa;
        #pragma unroll
        for (int i = 0; i < 8; ++i)
            pa.u[i] = *(const unsigned int*)(pb + n * 32 + kbase(i, half));

        // ---- O += P * V  (V^T rows contiguous in t-rel)
        #pragma unroll
        for (int nt = 0; nt < 4; ++nt) {
            v16bf vb = *(const v16bf*)(vs + (nt * 16 + n) * 32 + half * 16);
            o[nt] = wmma_bf16(pa.v, vb, o[nt]);
        }

        __syncthreads();                   // done reading buf before overwrite
    }

    // ---- finalize: O /= rowL, store bf16 into cat[B,S,H*DH]
    float inv[8];
    #pragma unroll
    for (int r = 0; r < 8; ++r) inv[r] = 1.0f / rowL[r];
    #pragma unroll
    for (int nt = 0; nt < 4; ++nt) {
        #pragma unroll
        for (int r = 0; r < 8; ++r) {
            int srow = s0 + r + 8 * half;
            cat[((size_t)(b * S_) + srow) * HD_ + h * DH_ + nt * 16 + n] =
                (__bf16)(o[nt][r] * inv[r]);
        }
    }
}

// ---------------------------------------------------------------------------
// Kernel 3: output projection. out[B*S, D] f32 = cat[B*S,1024]bf16 x W0t + b0
// Cat tile staged via async global->LDS DMA (ASYNCcnt).
// ---------------------------------------------------------------------------
__global__ __launch_bounds__(256)
void k_proj(const __bf16* __restrict__ cat, const __bf16* __restrict__ W0t,
            const float* __restrict__ b0, float* __restrict__ out) {
    __shared__ __bf16 cs[16 * HD_];     // 32 KB

    const int tid  = threadIdx.x;
    const int row0 = blockIdx.x * 16;   // flattened b*S + s

    #pragma unroll
    for (int it = 0; it < 8; ++it) {
        int idx8 = (tid + it * 256) * 8;
        async_copy_b128(cs + idx8, cat + (size_t)row0 * HD_ + idx8);
    }
    wait_asynccnt_0();
    __syncthreads();

    const int wave  = tid >> 5;
    const int lane  = tid & 31;
    const int n     = lane & 15;
    const int half  = lane >> 4;
    const int nbase = wave * 128;

    v8f acc[8];
    #pragma unroll
    for (int nt = 0; nt < 8; ++nt) {
        v8f z = {0.f,0.f,0.f,0.f,0.f,0.f,0.f,0.f};
        acc[nt] = z;
    }

    for (int kc = 0; kc < HD_ / 32; ++kc) {
        AFrag a;
        #pragma unroll
        for (int i = 0; i < 8; ++i)
            a.u[i] = *(const unsigned int*)(cs + n * HD_ + kc * 32 + kbase(i, half));
        #pragma unroll
        for (int nt = 0; nt < 8; ++nt) {
            int col = nbase + nt * 16 + n;
            v16bf bf = *(const v16bf*)(W0t + (size_t)col * HD_ + kc * 32 + half * 16);
            acc[nt] = wmma_bf16(a.v, bf, acc[nt]);
        }
    }

    #pragma unroll
    for (int nt = 0; nt < 8; ++nt) {
        int col = nbase + nt * 16 + n;
        float bb = b0[col];
        #pragma unroll
        for (int r = 0; r < 8; ++r)
            out[(size_t)(row0 + r + 8 * half) * D_ + col] = acc[nt][r] + bb;
    }
}

// ---------------------------------------------------------------------------
// Host launcher
// ---------------------------------------------------------------------------
extern "C" void kernel_launch(void* const* d_in, const int* in_sizes, int n_in,
                              void* d_out, int out_size, void* d_ws, size_t ws_size,
                              hipStream_t stream) {
    const float* x  = (const float*)d_in[0];
    const float* Wq = (const float*)d_in[1];
    const float* bq = (const float*)d_in[2];
    const float* Wk = (const float*)d_in[3];
    const float* bk = (const float*)d_in[4];
    const float* Wv = (const float*)d_in[5];
    const float* bv = (const float*)d_in[6];
    const float* W0 = (const float*)d_in[7];
    const float* b0 = (const float*)d_in[8];
    float* out = (float*)d_out;

    // Workspace layout (bytes)
    char* ws = (char*)d_ws;
    const size_t SZ_WT  = (size_t)H_ * DH_ * D_ * 2;      // 2 MB
    const size_t SZ_W0  = (size_t)HD_ * D_ * 2;           // 2 MB
    const size_t SZ_QKV = (size_t)B_ * H_ * S_ * DH_ * 2; // 16 MB

    __bf16* Wqt = (__bf16*)(ws);
    __bf16* Wkt = (__bf16*)(ws + SZ_WT);
    __bf16* Wvt = (__bf16*)(ws + 2 * SZ_WT);
    __bf16* W0t = (__bf16*)(ws + 3 * SZ_WT);
    __bf16* Qb  = (__bf16*)(ws + 3 * SZ_WT + SZ_W0);
    __bf16* Kb  = (__bf16*)(ws + 3 * SZ_WT + SZ_W0 + SZ_QKV);
    __bf16* Vtb = (__bf16*)(ws + 3 * SZ_WT + SZ_W0 + 2 * SZ_QKV);
    __bf16* cat = (__bf16*)(ws + 3 * SZ_WT + SZ_W0 + 3 * SZ_QKV);

    // Weight transposes: [H,D,DH]->[H,DH,D] and [HD,D]->[D,HD], f32->bf16
    {
        dim3 g((D_ * DH_ + 255) / 256, H_);
        k_prep_transpose<<<g, 256, 0, stream>>>(Wq, Wqt, D_, DH_);
        k_prep_transpose<<<g, 256, 0, stream>>>(Wk, Wkt, D_, DH_);
        k_prep_transpose<<<g, 256, 0, stream>>>(Wv, Wvt, D_, DH_);
        dim3 g0((HD_ * D_ + 255) / 256, 1);
        k_prep_transpose<<<g0, 256, 0, stream>>>(W0, W0t, HD_, D_);
    }

    // QKV projection: B*(S/16) blocks
    k_qkv<<<B_ * (S_ / 16), 256, 0, stream>>>(x, Wqt, Wkt, Wvt,
                                              bq, bk, bv, Qb, Kb, Vtb);

    // Flash attention: B*H*16 blocks (8 waves of 16 rows each)
    k_attn<<<B_ * H_ * 16, 256, 0, stream>>>(Qb, Kb, Vtb, cat);

    // Output projection: (B*S/16) blocks
    k_proj<<<(B_ * S_) / 16, 256, 0, stream>>>(cat, W0t, b0, out);
}
